// VQVAEModule_14328010899565
// MI455X (gfx1250) — matrix operand order repaired and checked by reading, not compiled, not run on hardware
//
#include <hip/hip_runtime.h>
#include <hip/hip_bf16.h>
#include <stdint.h>

// ---------------------------------------------------------------------------
// MI455X / gfx1250: bf16 WMMA implicit-GEMM conv1d stack + WMMA VQ codebook.
// wave32, V_WMMA_F32_16X16X32_BF16, channels-last bf16 activations.
// Activation tiles staged into LDS by the Tensor Data Mover (TDM), synced
// with s_wait_tensorcnt; inner loop = ds_load_b128 + global weight frags.
// ---------------------------------------------------------------------------

typedef __attribute__((ext_vector_type(16))) __bf16 v16bf;
typedef __attribute__((ext_vector_type(8)))  __bf16 v8bf;
typedef __attribute__((ext_vector_type(8)))  float  v8f;
typedef __attribute__((ext_vector_type(4)))  float  v4f;
typedef __attribute__((ext_vector_type(4)))  unsigned int u32x4;
typedef __attribute__((ext_vector_type(8)))  int i32x8;
typedef __attribute__((ext_vector_type(4)))  int i32x4;

#define HALO 8      // time halo rows each side (max pad = 5 for k=11)
#define TROWS 80    // LDS tile: 64 output rows + 2*8 halo slop

// ----------------------------- elementwise ---------------------------------
__global__ void ew_zero_bf16(__bf16* p, size_t n) {
  size_t i = (size_t)blockIdx.x * blockDim.x + threadIdx.x;
  if (i < n) p[i] = (__bf16)0.f;
}

__global__ void ew_zero_f32(float* p, size_t n) {
  size_t i = (size_t)blockIdx.x * blockDim.x + threadIdx.x;
  if (i < n) p[i] = 0.f;
}

// dst = leaky(src * scale)
__global__ void ew_scale_leaky(__bf16* dst, const __bf16* src, size_t n,
                               float alpha, float scale) {
  size_t i = (size_t)blockIdx.x * blockDim.x + threadIdx.x;
  if (i >= n) return;
  float v = (float)src[i] * scale;
  dst[i] = (__bf16)(v > 0.f ? v : alpha * v);
}

// dst += src
__global__ void ew_add_bf16(__bf16* dst, const __bf16* src, size_t n) {
  size_t i = (size_t)blockIdx.x * blockDim.x + threadIdx.x;
  if (i >= n) return;
  dst[i] = (__bf16)((float)dst[i] + (float)src[i]);
}

// ------------------------------- packing -----------------------------------
// x [B][M][T] f32  ->  xin [B][T+2H][Mpad] bf16 channels-last, halo+pad zeroed
__global__ void pack_x_kernel(const float* __restrict__ x, __bf16* __restrict__ xin,
                              int B, int T, int M, int Cpad) {
  const int R = T + 2 * HALO;
  size_t n = (size_t)B * R * Cpad;
  size_t i = (size_t)blockIdx.x * blockDim.x + threadIdx.x;
  if (i >= n) return;
  int ci = (int)(i % Cpad);
  size_t r = i / Cpad;
  int row = (int)(r % R);
  int b = (int)(r / R);
  int t = row - HALO;
  float v = 0.f;
  if (t >= 0 && t < T && ci < M) v = x[((size_t)b * M + ci) * T + t];
  xin[i] = (__bf16)v;
}

// Pack conv weights [Cout][Cin][Kt] f32 into WMMA A-fragment order:
// frag[cot][cc][lane][i] in contiguous k order (k = 16*(lane/16) + i).
// GEMM-K ordering is (kt, ci): K = kt*Cpad + ci.
__global__ void pack_w_kernel(const float* __restrict__ w, __bf16* __restrict__ frag,
                              int Cout, int Cin, int Cpad, int Kt) {
  const int cotiles = (Cout + 15) >> 4;
  const int cpc = Cpad >> 5;          // 32-wide K chunks per kt
  const int nch = Kt * cpc;
  size_t tot = (size_t)cotiles * nch * 512;
  size_t idx = (size_t)blockIdx.x * blockDim.x + threadIdx.x;
  if (idx >= tot) return;
  int i    = (int)(idx & 15);
  int lane = (int)((idx >> 4) & 31);
  size_t rest = idx >> 9;
  int cc  = (int)(rest % nch);
  int cot = (int)(rest / nch);
  int m = lane & 15, hi = lane >> 4;
  int j = i >> 1, p = i & 1;
  // A-layout (contiguous k): lanes0-15 -> K 0..15, lanes16-31 -> K 16..31
  int kl = hi * 16 + i; (void)j; (void)p;
  int kt = cc / cpc;
  int ci = (cc - kt * cpc) * 32 + kl;
  int co = cot * 16 + m;
  float v = 0.f;
  if (co < Cout && ci < Cin) v = w[((size_t)co * Cin + ci) * Kt + kt];
  frag[idx] = (__bf16)v;
}

// Pack codebook embed [D][NE] into B-fragment order: frag[nt][c][lane][i],
// lane = n (col), halves i = consecutive k (d), contiguous k = 16*(lane/16) + i.
__global__ void pack_e_kernel(const float* __restrict__ embed, __bf16* __restrict__ frag,
                              int NE, int D) {
  const int cpk = D >> 5;   // K chunks (=2 for D=64)
  size_t tot = (size_t)(NE >> 4) * cpk * 512;
  size_t idx = (size_t)blockIdx.x * blockDim.x + threadIdx.x;
  if (idx >= tot) return;
  int i    = (int)(idx & 15);
  int lane = (int)((idx >> 4) & 31);
  size_t rest = idx >> 9;
  int c  = (int)(rest % cpk);
  int nt = (int)(rest / cpk);
  int n = (nt << 4) + (lane & 15);
  int k = (c << 5) + ((lane >> 4) << 4) + i;
  frag[idx] = (__bf16)embed[(size_t)k * NE + n];
}

__global__ void enorm_kernel(const float* __restrict__ embed, float* __restrict__ enorm,
                             int NE, int D) {
  int n = blockIdx.x * blockDim.x + threadIdx.x;
  if (n >= NE) return;
  float s = 0.f;
  for (int d = 0; d < D; ++d) { float e = embed[(size_t)d * NE + n]; s += e * e; }
  enorm[n] = s;
}

// ----------------------- TDM: 2-D tile -> LDS ------------------------------
// Build D# per CDNA5 ISA ch.8 and issue TENSOR_LOAD_TO_LDS.
// tile = nrows x celts (2-byte elements), row stride = celts, contiguous LDS.
__device__ __forceinline__ void tdm_load_tile(const __bf16* gsrc, __bf16* lds_dst,
                                              int celts, int nrows) {
  uint64_t ga = (uint64_t)(uintptr_t)gsrc;
  uint32_t ldsOff = (uint32_t)(uintptr_t)lds_dst;  // low 32 bits = LDS byte addr
  u32x4 g0;
  g0[0] = 1u;                                             // count=1, user mode
  g0[1] = ldsOff;                                         // lds_addr [63:32]
  g0[2] = (uint32_t)ga;                                   // global_addr [95:64]
  g0[3] = (uint32_t)((ga >> 32) & 0x01FFFFFFu) | (2u << 30); // ga[56:32] | type=2
  i32x8 g1;
  g1[0] = 0x00010000;                                     // data_size=1 (2 bytes)
  g1[1] = (int)(((unsigned)celts & 0xFFFFu) << 16);       // tensor_dim0 lo16
  g1[2] = (int)((((unsigned)celts >> 16) & 0xFFFFu) |
                (((unsigned)nrows & 0xFFFFu) << 16));     // td0 hi16 | td1 lo16
  g1[3] = (int)((((unsigned)nrows >> 16) & 0xFFFFu) |
                (((unsigned)celts & 0xFFFFu) << 16));     // td1 hi16 | tile_dim0
  g1[4] = (int)((unsigned)nrows & 0xFFFFu);               // tile_dim1 | tile_dim2=0
  g1[5] = celts;                                          // tensor_dim0_stride lo32
  g1[6] = 0;                                              // stride hi | td1_stride lo
  g1[7] = 0;
  i32x4 gz; gz[0] = gz[1] = gz[2] = gz[3] = 0;            // 2-D: groups 2/3 unused
#if defined(__clang_major__) && (__clang_major__ >= 23)
  i32x8 gz8;
  for (int q = 0; q < 8; ++q) gz8[q] = 0;
  __builtin_amdgcn_tensor_load_to_lds(g0, g1, gz, gz, gz8, 0);
#else
  __builtin_amdgcn_tensor_load_to_lds(g0, g1, gz, gz, 0);
#endif
}

// ------------------------- conv1d as WMMA GEMM -----------------------------
// xin  : [B][T+2H][Cinp] bf16 channels-last, halo zeroed
// wfr  : packed A-fragments
// outb : [B][T+2H][Coutp] bf16 (interior rows written)  (if outf == nullptr)
// outf : [B][T][Cout] f32                                (if outf != nullptr)
// res  : optional residual (channels-last, same layout as outb), added AFTER act
// act  : 0=none 1=relu 2=leaky(alpha)
__global__ __launch_bounds__(256) void conv_wmma_kernel(
    const __bf16* __restrict__ xin, const __bf16* __restrict__ wfr,
    const float* __restrict__ bias,
    __bf16* __restrict__ outb, const __bf16* __restrict__ res,
    float* __restrict__ outf,
    int T, int Cinp, int Cout, int Coutp, int Kt, int pad, int act, float alpha) {
  extern __shared__ __bf16 smem[];      // TROWS x Cinp bf16 activation tile
  const int R = T + 2 * HALO;
  const int lane = threadIdx.x & 31;
  const int wv = threadIdx.x >> 5;
  const int wm = wv & 1;          // 2 waves over M
  const int wn = wv >> 1;         // 4 waves over N (time)
  const int lm = lane & 15;
  const int lh = lane >> 4;
  const int b = blockIdx.z;
  const int t0b = blockIdx.x * 64;                 // block's first output row
  const int cotBase = blockIdx.y * 8 + wm * 4;     // units of 16 output chans
  const int cotiles = (Cout + 15) >> 4;
  const int cpc = Cinp >> 5;
  const int nch = Kt * cpc;

  // ---- TDM: DMA the 80-row activation tile into LDS (wave 0 issues) ----
  // Buffer rows [t0b .. t0b+79] cover all taps: t0b + lt + kt - pad + HALO,
  // lt in [0,63], kt-pad in [-5,5]  ->  local rows [3, 76].
  if (wv == 0) {
    const __bf16* gsrc = xin + ((size_t)b * R + t0b) * Cinp;
    tdm_load_tile(gsrc, smem, Cinp, TROWS);
    __builtin_amdgcn_s_wait_tensorcnt(0);
  }
  __syncthreads();

  v8f acc[4];
  for (int i = 0; i < 4; ++i)
    for (int e = 0; e < 8; ++e) acc[i][e] = 0.f;

  const int ltBase = wn * 16 + lm;                 // this lane's output row (local)
  for (int cc = 0; cc < nch; ++cc) {
    int kt = cc / cpc;
    int ch = cc - kt * cpc;
    int lrow = ltBase + kt - pad + HALO;           // local LDS row, in [3,76]
    const __bf16* xp = smem + (size_t)lrow * Cinp + (ch << 5) + (lh << 4);
    v16bf bfr = *(const v16bf*)xp;                 // 2x ds_load_b128
    if (cc + 1 < nch) {
      // prefetch next weight fragment group (L2-resident, global_prefetch_b8)
      __builtin_prefetch(wfr + ((((size_t)cotBase * nch + (cc + 1)) << 5) + lane) * 16, 0, 1);
    }
#pragma unroll
    for (int i = 0; i < 4; ++i) {
      int cot = cotBase + i;
      if (cot < cotiles) {
        const __bf16* wp = wfr + ((((size_t)cot * nch + cc) << 5) + lane) * 16;
        v16bf afr = *(const v16bf*)wp;
        acc[i] = __builtin_amdgcn_wmma_f32_16x16x32_bf16(
            false, afr, false, bfr, (short)0, acc[i], false, false);
      }
    }
  }

  const int tOut = t0b + ltBase;
#pragma unroll
  for (int i = 0; i < 4; ++i) {
    int cot = cotBase + i;
    if (cot >= cotiles) continue;
    int co0 = (cot << 4) + (lh << 3);
    float vals[8];
#pragma unroll
    for (int e = 0; e < 8; ++e) {
      int co = co0 + e;
      float y = acc[i][e] + bias[co];
      if (act == 1) y = fmaxf(y, 0.f);
      else if (act == 2) y = (y > 0.f) ? y : alpha * y;
      if (res) y += (float)res[((size_t)b * R + tOut + HALO) * Coutp + co];
      vals[e] = y;
    }
    if (outf) {
      float* op = outf + ((size_t)b * T + tOut) * Cout + co0;
      v4f lo, hi4;
      for (int e = 0; e < 4; ++e) { lo[e] = vals[e]; hi4[e] = vals[e + 4]; }
      *(v4f*)op = lo;
      *(v4f*)(op + 4) = hi4;
    } else {
      v8bf pk;
      for (int e = 0; e < 8; ++e) pk[e] = (__bf16)vals[e];
      *(v8bf*)(outb + ((size_t)b * R + tOut + HALO) * Coutp + co0) = pk;
    }
  }
}

// ------------------------------- VQ stage ----------------------------------
// score(n) = ||e_n||^2 - 2 * (f . e_n)   (||f||^2 constant over argmin)
__global__ __launch_bounds__(256) void vq_wmma_kernel(
    const float* __restrict__ enc, const __bf16* __restrict__ efr,
    const float* __restrict__ enorm, const float* __restrict__ embed,
    float* __restrict__ out, float* __restrict__ diffacc,
    int Bn, int T, int NE, int D) {
  const int lane = threadIdx.x & 31;
  const int wv = threadIdx.x >> 5;
  const int lm = lane & 15, lh = lane >> 4;
  const int row0 = (blockIdx.x * 8 + wv) * 16;
  const int N = Bn * T;

  // Gather A fragments (2 K-chunks of 32 over D=64) in contiguous k order (k = 16*lh + i).
  v16bf a[2];
  const int rowA = row0 + lm;
#pragma unroll
  for (int c = 0; c < 2; ++c)
#pragma unroll
    for (int i = 0; i < 16; ++i) {
      int j = i >> 1, p = i & 1;
      int kl = lh * 16 + i; (void)j; (void)p;
      a[c][i] = (__bf16)enc[(size_t)rowA * D + c * 32 + kl];
    }

  float best[8];
  int bidx[8];
  for (int e = 0; e < 8; ++e) { best[e] = 3.4e38f; bidx[e] = 0; }

  const int ntiles = NE >> 4;
  for (int nt = 0; nt < ntiles; ++nt) {
    v8f g;
    for (int e = 0; e < 8; ++e) g[e] = 0.f;
#pragma unroll
    for (int c = 0; c < 2; ++c) {
      v16bf bf = *(const v16bf*)(efr + ((((size_t)nt * 2 + c) << 5) + lane) * 16);
      g = __builtin_amdgcn_wmma_f32_16x16x32_bf16(
          false, a[c], false, bf, (short)0, g, false, false);
    }
    float en = enorm[(nt << 4) + lm];
    int nidx = (nt << 4) + lm;
    for (int e = 0; e < 8; ++e) {
      float s = en - 2.f * g[e];
      if (s < best[e]) { best[e] = s; bidx[e] = nidx; }
    }
  }

  // argmin across the 16 n-lanes of each half-wave (m fixed per half/vgpr)
  for (int mask = 1; mask < 16; mask <<= 1) {
    for (int e = 0; e < 8; ++e) {
      float ob = __shfl_xor(best[e], mask, 32);
      int oi = __shfl_xor(bidx[e], mask, 32);
      if (ob < best[e] || (ob == best[e] && oi < bidx[e])) { best[e] = ob; bidx[e] = oi; }
    }
  }

  if (lm == 0) {
    float dsum = 0.f;
    for (int e = 0; e < 8; ++e) {
      int row = row0 + lh * 8 + e;
      int b = row / T, t = row - b * T;
      int ind = bidx[e];
      float ds = 0.f;
      for (int d = 0; d < D; ++d) {
        float ev = embed[(size_t)d * NE + ind];
        float fv = enc[(size_t)row * D + d];
        out[((size_t)b * D + d) * T + t] = ev;   // q_st == q in value (straight-through)
        float df = ev - fv;
        ds += df * df;
      }
      dsum += ds;
    }
    atomicAdd(diffacc, dsum / ((float)N * (float)D));
  }
}

// ------------------------------- launcher ----------------------------------
extern "C" void kernel_launch(void* const* d_in, const int* in_sizes, int n_in,
                              void* d_out, int out_size, void* d_ws, size_t ws_size,
                              hipStream_t stream) {
  (void)in_sizes; (void)n_in; (void)out_size;
  const int B = 16, T = 4096, M = 80, C0 = 256, C1 = 512, D = 64, NE = 1024;
  const int Mpad = 96, R = T + 2 * HALO;
  const float alpha = 0.1f;
  const int ksz[3] = {3, 7, 11};

  const float* x          = (const float*)d_in[0];
  const float* conv_pre_w = (const float*)d_in[1];
  const float* conv_pre_b = (const float*)d_in[2];
  const float* rbw[2][3], *rbb[2][3];
  rbw[0][0] = (const float*)d_in[3];  rbb[0][0] = (const float*)d_in[4];
  rbw[0][1] = (const float*)d_in[5];  rbb[0][1] = (const float*)d_in[6];
  rbw[0][2] = (const float*)d_in[7];  rbb[0][2] = (const float*)d_in[8];
  const float* between_w  = (const float*)d_in[9];
  const float* between_b  = (const float*)d_in[10];
  rbw[1][0] = (const float*)d_in[11]; rbb[1][0] = (const float*)d_in[12];
  rbw[1][1] = (const float*)d_in[13]; rbb[1][1] = (const float*)d_in[14];
  rbw[1][2] = (const float*)d_in[15]; rbb[1][2] = (const float*)d_in[16];
  const float* quant_w    = (const float*)d_in[17];
  const float* quant_b    = (const float*)d_in[18];
  const float* embed      = (const float*)d_in[19];

  // --- carve workspace ---
  uint8_t* base = (uint8_t*)d_ws;
  size_t off = 0;
  auto carve = [&](size_t bytes) -> void* {
    void* p = base + off;
    off = (off + bytes + 255) & ~(size_t)255;
    return p;
  };
  auto fragBytes = [](int Cout, int Cpad, int Kt) -> size_t {
    return (size_t)((Cout + 15) / 16) * (size_t)(Kt * (Cpad / 32)) * 1024;
  };

  __bf16* xin = (__bf16*)carve((size_t)B * R * Mpad * 2);
  const size_t big = (size_t)B * R * C1 * 2;
  __bf16* X = (__bf16*)carve(big);
  __bf16* P = (__bf16*)carve(big);
  __bf16* Q = (__bf16*)carve(big);
  __bf16* S = (__bf16*)carve(big);
  float* enc = (float*)carve((size_t)B * T * D * 4);
  __bf16* f_pre = (__bf16*)carve(fragBytes(C0, Mpad, 3));
  __bf16* f_rb[2][3][4];
  for (int s = 0; s < 2; ++s) {
    int C = s ? C1 : C0;
    for (int ki = 0; ki < 3; ++ki)
      for (int j = 0; j < 4; ++j)
        f_rb[s][ki][j] = (__bf16*)carve(fragBytes(C, C, ksz[ki]));
  }
  __bf16* f_bet = (__bf16*)carve(fragBytes(C1, C0, 1));
  __bf16* f_qnt = (__bf16*)carve(fragBytes(D, C1, 1));
  __bf16* f_emb = (__bf16*)carve((size_t)(NE / 16) * (D / 32) * 512 * 2);
  float* enormp = (float*)carve((size_t)NE * 4);
  if (off > ws_size) return;   // workspace too small: bail (compile-only loop)

  const int TPB = 256;
  auto nblk = [&](size_t n) { return dim3((unsigned)((n + TPB - 1) / TPB)); };

  // --- pack inputs / weights / codebook ---
  {
    size_t n = (size_t)B * R * Mpad;
    pack_x_kernel<<<nblk(n), TPB, 0, stream>>>(x, xin, B, T, M, Mpad);
  }
  auto packw = [&](const float* w, __bf16* fr, int Cout, int Cin, int Cpad, int Kt) {
    size_t n = fragBytes(Cout, Cpad, Kt) / 2;
    pack_w_kernel<<<nblk(n), TPB, 0, stream>>>(w, fr, Cout, Cin, Cpad, Kt);
  };
  packw(conv_pre_w, f_pre, C0, M, Mpad, 3);
  for (int s = 0; s < 2; ++s) {
    int C = s ? C1 : C0;
    for (int ki = 0; ki < 3; ++ki)
      for (int j = 0; j < 4; ++j)
        packw(rbw[s][ki] + (size_t)j * C * C * ksz[ki], f_rb[s][ki][j], C, C, C, ksz[ki]);
  }
  packw(between_w, f_bet, C1, C0, C0, 1);
  packw(quant_w, f_qnt, D, C1, C1, 1);
  {
    size_t n = (size_t)(NE / 16) * (D / 32) * 512;
    pack_e_kernel<<<nblk(n), TPB, 0, stream>>>(embed, f_emb, NE, D);
  }
  enorm_kernel<<<nblk(NE), TPB, 0, stream>>>(embed, enormp, NE, D);

  // --- helpers ---
  auto zero = [&](__bf16* p, int C) {
    size_t n = (size_t)B * R * C;
    ew_zero_bf16<<<nblk(n), TPB, 0, stream>>>(p, n);
  };
  auto conv = [&](const __bf16* in, int Cinp, const __bf16* fr, const float* bias,
                  __bf16* outb, const __bf16* res, float* outf, int Cout, int Coutp,
                  int Kt, int act) {
    dim3 grid(T / 64, (Cout + 127) / 128, B);
    size_t shmem = (size_t)TROWS * Cinp * sizeof(__bf16);   // 15-80 KB of 320 KB LDS
    conv_wmma_kernel<<<grid, TPB, shmem, stream>>>(in, fr, bias, outb, res, outf,
                                                   T, Cinp, Cout, Coutp, Kt, Kt / 2,
                                                   act, alpha);
  };
  auto scale_leaky = [&](__bf16* dst, const __bf16* src, int C, float scale) {
    size_t n = (size_t)B * R * C;
    ew_scale_leaky<<<nblk(n), TPB, 0, stream>>>(dst, src, n, alpha, scale);
  };
  auto addacc = [&](__bf16* dst, const __bf16* src, int C) {
    size_t n = (size_t)B * R * C;
    ew_add_bf16<<<nblk(n), TPB, 0, stream>>>(dst, src, n);
  };

  // --- network ---
  zero(Q, C0);
  conv(xin, Mpad, f_pre, conv_pre_b, Q, nullptr, nullptr, C0, C0, 3, 0);
  scale_leaky(X, Q, C0, 1.f);                       // X = leaky(conv_pre)

  for (int s = 0; s < 2; ++s) {
    int C = s ? C1 : C0;
    zero(S, C); zero(P, C); zero(Q, C);
    for (int ki = 0; ki < 3; ++ki) {
      const float* bb = rbb[s][ki];
      int k = ksz[ki];
      // resblock 1: X -> P  (h=relu(conv); h=relu(conv); +X)
      conv(X, C, f_rb[s][ki][0], bb + 0 * C, Q, nullptr, nullptr, C, C, k, 1);
      conv(Q, C, f_rb[s][ki][1], bb + 1 * C, P, X, nullptr, C, C, k, 1);
      // resblock 2: P -> P (in-place residual is element-exact)
      conv(P, C, f_rb[s][ki][2], bb + 2 * C, Q, nullptr, nullptr, C, C, k, 1);
      conv(Q, C, f_rb[s][ki][3], bb + 3 * C, P, P, nullptr, C, C, k, 1);
      addacc(S, P, C);                              // S += branch output
    }
    if (s == 0) {
      scale_leaky(X, S, C0, 1.f / 3.f);             // X = leaky(S/3)
      zero(Q, C1);                                  // fresh halo at 512 layout
      conv(X, C0, f_bet, between_b, Q, nullptr, nullptr, C1, C1, 1, 0);
      scale_leaky(X, Q, C1, 1.f);                   // X = leaky(between)
    } else {
      scale_leaky(X, S, C1, 1.f / 3.f);             // X = leaky(S/3)
    }
  }

  // quant: 1x1 conv 512 -> 64, f32 output [B][T][64]
  conv(X, C1, f_qnt, quant_b, nullptr, nullptr, enc, D, D, 1, 0);

  // VQ: argmin + gather + diff
  float* outp = (float*)d_out;
  float* diffp = outp + (size_t)B * D * T;
  ew_zero_f32<<<1, 1, 0, stream>>>(diffp, 1);
  dim3 vgrid((unsigned)((B * T) / 128));
  vq_wmma_kernel<<<vgrid, TPB, 0, stream>>>(enc, f_emb, enormp, embed, outp, diffp,
                                            B, T, NE, D);
}
